// SequenceClassifierCNN_80315888435789
// MI455X (gfx1250) — matrix operand order repaired and checked by reading, not compiled, not run on hardware
//
#include <hip/hip_runtime.h>
#include <cstdint>
#include <cstddef>

// ---------------------------------------------------------------------------
// Types / WMMA helpers (CDNA5 gfx1250, wave32)
// ---------------------------------------------------------------------------
typedef __attribute__((ext_vector_type(16))) _Float16 v16h;
typedef __attribute__((ext_vector_type(8)))  float    v8f;

union FragU { uint4 u[2]; v16h h; };

// A fragment: 16x32 (MxK) f16 tile, row-major, leading dim ld (in halves).
// Lanes 0-15: rows 0-15, K=0..7 & 16..23 ; lanes 16-31: rows 0-15, K=8..15 & 24..31.
__device__ __forceinline__ v16h load_fragA(const _Float16* tile, int ld, int lane) {
  const int r  = lane & 15;
  const int ko = (lane >> 4) << 3;            // 0 or 8
  const _Float16* p = tile + r * ld + ko;
  FragU f;
  f.u[0] = *(const uint4*)(p);
  f.u[1] = *(const uint4*)(p + 16);
  return f.h;
}

// B fragment: 32x16 (KxN) = W^T where W is [N][K] row-major (ld = row stride in halves).
// Lanes 0-15: col n=lane, K=0..7 & 16..23 ; lanes 16-31: col n=lane-16, K=8..15 & 24..31.
__device__ __forceinline__ v16h load_fragB(const _Float16* wt, int ld, int lane) {
  const int n  = lane & 15;
  const int ko = (lane >> 4) << 3;            // 0 or 8
  const _Float16* p = wt + n * ld + ko;
  FragU f;
  f.u[0] = *(const uint4*)(p);
  f.u[1] = *(const uint4*)(p + 16);
  return f.h;
}

__device__ __forceinline__ v8f wmma_f16(v16h a, v16h b, v8f c) {
  // 8 args: (neg_a, A, neg_b, B, c_mod, C, reuse_a, reuse_b)
  return __builtin_amdgcn_wmma_f32_16x16x32_f16(false, a, false, b, (short)0, c,
                                                false, false);
}

// Fast activations: hardware v_rcp_f32 / v_tanh_f32 instead of IEEE div chains.
__device__ __forceinline__ float fast_rcp(float x) {
#if __has_builtin(__builtin_amdgcn_rcpf)
  return __builtin_amdgcn_rcpf(x);
#else
  return 1.0f / x;
#endif
}
__device__ __forceinline__ float sigm(float x) {
  return fast_rcp(1.0f + __expf(-x));     // e->0: 1 ; e->inf: rcp(inf)=0
}
__device__ __forceinline__ float tanh_f(float x) {
#if __has_builtin(__builtin_amdgcn_tanhf)
  return __builtin_amdgcn_tanhf(x);
#else
  float ax = fabsf(x);
  float e  = __expf(-2.0f * ax);          // in (0,1], no overflow
  float t  = (1.0f - e) * fast_rcp(1.0f + e);
  return copysignf(t, x);
#endif
}

// ---------------------------------------------------------------------------
// Prep kernels: f32 -> padded f16 weights (optionally bias in column K), bias sums
// ---------------------------------------------------------------------------
__global__ void cvt_pad_k(const float* __restrict__ src, _Float16* __restrict__ dst,
                          int N, int K, int KPAD) {
  int i = blockIdx.x * 256 + threadIdx.x;
  if (i >= N * KPAD) return;
  int n = i / KPAD, k = i - n * KPAD;
  dst[i] = (k < K) ? (_Float16)src[(size_t)n * K + k] : (_Float16)0.0f;
}

// Same, but column K carries the (f32-summed) bias so the GEMM against a
// constant-1.0 activation column applies the bias for free.
__global__ void cvt_pad_bias_k(const float* __restrict__ src,
                               const float* __restrict__ bias,
                               _Float16* __restrict__ dst,
                               int N, int K, int KPAD) {
  int i = blockIdx.x * 256 + threadIdx.x;
  if (i >= N * KPAD) return;
  int n = i / KPAD, k = i - n * KPAD;
  _Float16 v = (_Float16)0.0f;
  if (k < K)       v = (_Float16)src[(size_t)n * K + k];
  else if (k == K) v = (_Float16)bias[n];
  dst[i] = v;
}

__global__ void bias_sum_k(const float* __restrict__ a, const float* __restrict__ b,
                           float* __restrict__ o, int n) {
  int i = blockIdx.x * 256 + threadIdx.x;
  if (i < n) o[i] = a[i] + b[i];
}

// ---------------------------------------------------------------------------
// Fused LSTM layer. One block owns a batch tile, loops over T internally.
// Gate order (PyTorch): i, f, g, o at rows [0,H),[H,2H),[2H,3H),[3H,4H) of W.
// Wave w: mg = w/NT (batch sub-tile), j = w%NT (hidden column tile); owns gate
// tiles j, j+NT, j+2NT, j+3NT -> c/h update stays entirely in registers.
// All weight B-fragments pinned in VGPRs; accumulators start at inline-0 SRC2.
// x_{t+1} prefetched via precomputed 32-bit per-thread offsets (no per-step
// address math) into a double-buffered LDS tile; 2 barriers per step.
// BIASCOL: bias folded into weight column INP against a constant-1.0 x column
// (and the post-GEMM bias add is compiled out entirely).
// ---------------------------------------------------------------------------
template<int INP, int INPAD, int HID, bool F32IN, bool BIASCOL>
__global__ __launch_bounds__(256) void lstm_layer_k(
    const void* __restrict__ xin,          // [B,T,INP] f32 (layer1) or f16
    const _Float16* __restrict__ Wih,      // [4H][INPAD] f16, zero-padded
    const _Float16* __restrict__ Whh,      // [4H][HID]   f16
    const float* __restrict__ bias,        // [4H] = b_ih + b_hh
    _Float16* __restrict__ hout,           // [B,T,HID] relu(h) f16
    int T)
{
  constexpr int NT  = HID / 16;            // column tiles per gate (4 or 8)
  constexpr int MG  = 128 / HID;           // batch sub-tiles per block (2 or 1)
  constexpr int BT  = 16 * MG;             // batch rows per block (32 or 16)
  constexpr int XKT = INPAD / 32;          // K-tiles, input part
  constexpr int HKT = HID / 32;            // K-tiles, hidden part
  constexpr int NE  = BT * INP;            // scalar elements per x tile (f32 path)
  constexpr int NP  = (NE + 255) / 256;    // staged f32 elements per thread
  constexpr int XC  = INP / 8;             // 16B chunks per row (f16 path)
  constexpr int NCH = (BT * XC + 255) / 256; // staged chunks per thread
  constexpr int NST = F32IN ? NP : NCH;    // staged slots per thread

  static_assert(BT * (HID / 8) == 256, "one hout chunk per thread");

  __shared__ alignas(16) _Float16 xs[2][BT * INPAD]; // double-buffered x tile
  __shared__ alignas(16) _Float16 hs[BT * HID];      // h_{t-1} tile (f16)

  const int tid  = threadIdx.x;
  const int lane = tid & 31;
  const int wave = tid >> 5;
  const int mg   = wave / NT;
  const int j    = wave % NT;
  const int b0   = blockIdx.x * BT;

  const float*    xf = (const float*)xin;
  const _Float16* xh = (const _Float16*)xin;

  // ---- pin weight B-fragments in registers (loaded once) ----
  v16h wx[XKT][4];
  v16h wh[HKT][4];
  float bsr[4] = {0.0f, 0.0f, 0.0f, 0.0f};
#pragma unroll
  for (int g = 0; g < 4; ++g) {
    const int n16 = (g * NT + j) * 16;
#pragma unroll
    for (int kt = 0; kt < XKT; ++kt)
      wx[kt][g] = load_fragB(Wih + (size_t)n16 * INPAD + kt * 32, INPAD, lane);
#pragma unroll
    for (int kt = 0; kt < HKT; ++kt)
      wh[kt][g] = load_fragB(Whh + (size_t)n16 * HID + kt * 32, HID, lane);
    if (!BIASCOL) bsr[g] = bias[n16 + (lane & 15)];
  }

  // ---- precomputed per-thread offsets (t-invariant) ----
  int goff[NST];   // global element offset sans the tt*INP term
  int loff[NST];   // LDS half offset within one x buffer
  if (F32IN) {
#pragma unroll
    for (int u = 0; u < NP; ++u) {
      int idx  = tid + u * 256;
      int safe = (idx < NE) ? idx : 0;
      int r = safe / INP, k = safe - r * INP;
      goff[u] = (b0 + r) * (T * INP) + k;
      loff[u] = r * INPAD + k;
    }
  } else {
#pragma unroll
    for (int u = 0; u < NCH; ++u) {
      int cidx = tid + u * 256;
      int safe = (cidx < BT * XC) ? cidx : 0;
      int r = safe / XC, cc = safe - r * XC;
      goff[u] = (b0 + r) * (T * INP) + cc * 8;
      loff[u] = r * INPAD + cc * 8;
    }
  }
  // hout chunk (exactly one per thread)
  constexpr int CH = HID / 8;
  const int hr = tid / CH, hc = tid - hr * CH;
  const int hout_base = (b0 + hr) * (T * HID) + hc * 8;  // element offset
  const int hlds_off  = hr * HID + hc * 8;

  // ---- zero xs (pad columns must stay 0: NaN*0=NaN) and hs (h0 = 0) ----
  for (int i = tid; i < BT * INPAD; i += 256)
    ((unsigned int*)xs)[i] = 0u;                 // covers both buffers (2x halves)
  for (int i = tid; i < (BT * HID) / 2; i += 256) ((unsigned int*)hs)[i] = 0u;

  float creg[8];
#pragma unroll
  for (int e = 0; e < 8; ++e) creg[e] = 0.0f;

  __syncthreads();   // zero-fill done before constant/staging writes

  // ---- register staging for x prefetch ----
  float pre_f[F32IN ? NP : 1];
  uint4 pre_h[F32IN ? 1 : NCH];

  auto stage_x = [&](int tt) {
    if (F32IN) {
      const float* xf_t = xf + (size_t)tt * INP;   // uniform slice pointer
#pragma unroll
      for (int u = 0; u < NP; ++u) pre_f[u] = xf_t[goff[u]];
    } else {
      const _Float16* xh_t = xh + (size_t)tt * INP;
#pragma unroll
      for (int u = 0; u < NCH; ++u)
        pre_h[u] = *(const uint4*)(xh_t + goff[u]);
    }
  };
  auto commit_x = [&](int buf) {
    if (F32IN) {
#pragma unroll
      for (int u = 0; u < NP; ++u)
        if (tid + u * 256 < NE) xs[buf][loff[u]] = (_Float16)pre_f[u];
    } else {
#pragma unroll
      for (int u = 0; u < NCH; ++u)
        if (tid + u * 256 < BT * XC) *(uint4*)&xs[buf][loff[u]] = pre_h[u];
    }
  };

  // prologue: constant-1.0 bias column (both buffers) + x_0 into buffer 0
  if (BIASCOL) {
    if (tid < 2 * BT) {
      int bb = tid >> (MG == 2 ? 5 : 4);   // buffer index
      int rr = tid & (BT - 1);
      xs[bb][rr * INPAD + INP] = (_Float16)1.0f;
    }
  }
  stage_x(0);
  commit_x(0);
  __syncthreads();

  const int col   = j * 16 + (lane & 15);         // hidden column owned by lane
  const int rbase = ((lane >> 4) << 3) + mg * 16; // first D-matrix row of lane

  for (int t = 0; t < T; ++t) {
    const int buf = t & 1;

    // -- 1. issue global loads for x_{t+1} (latency overlapped with WMMAs) --
    stage_x((t + 1 < T) ? (t + 1) : t);

    // -- 2. gate GEMMs from inline-0 accumulators --
    v8f acc[4] = {};
#pragma unroll
    for (int kt = 0; kt < XKT; ++kt) {
      v16h a = load_fragA(&xs[buf][mg * 16 * INPAD + kt * 32], INPAD, lane);
#pragma unroll
      for (int g = 0; g < 4; ++g) acc[g] = wmma_f16(a, wx[kt][g], acc[g]);
    }
#pragma unroll
    for (int kt = 0; kt < HKT; ++kt) {
      v16h a = load_fragA(&hs[mg * 16 * HID + kt * 32], HID, lane);
#pragma unroll
      for (int g = 0; g < 4; ++g) acc[g] = wmma_f16(a, wh[kt][g], acc[g]);
    }

    // -- 3. LSTM cell update (bias add compiled out when BIASCOL) --
    float hv[8];
#pragma unroll
    for (int e = 0; e < 8; ++e) {
      float a0 = BIASCOL ? acc[0][e] : (acc[0][e] + bsr[0]);
      float a1 = BIASCOL ? acc[1][e] : (acc[1][e] + bsr[1]);
      float a2 = BIASCOL ? acc[2][e] : (acc[2][e] + bsr[2]);
      float a3 = BIASCOL ? acc[3][e] : (acc[3][e] + bsr[3]);
      float ig = sigm(a0);
      float fg = sigm(a1);
      float gg = tanh_f(a2);
      float og = sigm(a3);
      float cv = fg * creg[e] + ig * gg;
      creg[e] = cv;
      hv[e]   = og * tanh_f(cv);
    }

    __syncthreads();   // all waves finished reading hs and xs[buf^1] (t-1)

    // -- 4. scatter h_t into LDS; commit prefetched x_{t+1} to other buffer --
#pragma unroll
    for (int e = 0; e < 8; ++e)
      hs[(rbase + e) * HID + col] = (_Float16)hv[e];
    commit_x(buf ^ 1);

    __syncthreads();   // hs = h_t and xs[buf^1] = x_{t+1} now complete

    // -- 5. relu(h_t) -> global f16 [B,T,HID]; one 16B chunk per thread --
    {
      uint4 v = *(const uint4*)&hs[hlds_off];
      _Float16* hp = (_Float16*)&v;
#pragma unroll
      for (int q = 0; q < 8; ++q)
        hp[q] = (hp[q] > (_Float16)0.0f) ? hp[q] : (_Float16)0.0f;
      *(uint4*)(hout + (size_t)hout_base + (size_t)t * HID) = v;
    }
  }
}

// ---------------------------------------------------------------------------
// Classifier head: relu already folded into h3; FC 64->64->32->10 + softmax.
// One thread per batch row; weights staged in LDS.
// ---------------------------------------------------------------------------
__global__ __launch_bounds__(256) void head_k(
    const _Float16* __restrict__ h3,    // [B,T,64] relu'd f16
    const float* __restrict__ w1, const float* __restrict__ b1,
    const float* __restrict__ w2, const float* __restrict__ b2,
    const float* __restrict__ w3, const float* __restrict__ b3,
    float* __restrict__ out, int T)
{
  __shared__ float s1[64 * 64], sb1[64];
  __shared__ float s2[32 * 64], sb2[32];
  __shared__ float s3[10 * 32], sb3[10];
  const int tid = threadIdx.x;
  for (int i = tid; i < 64 * 64; i += 256) s1[i] = w1[i];
  for (int i = tid; i < 64;      i += 256) sb1[i] = b1[i];
  for (int i = tid; i < 32 * 64; i += 256) s2[i] = w2[i];
  for (int i = tid; i < 32;      i += 256) sb2[i] = b2[i];
  for (int i = tid; i < 10 * 32; i += 256) s3[i] = w3[i];
  for (int i = tid; i < 10;      i += 256) sb3[i] = b3[i];
  __syncthreads();

  const int b = blockIdx.x * 256 + tid;
  const _Float16* xp = h3 + ((size_t)b * T + (T - 1)) * 64;
  float x[64];
#pragma unroll
  for (int k = 0; k < 64; ++k) x[k] = (float)xp[k];

  float h1v[64];
  for (int o = 0; o < 64; ++o) {
    float s = sb1[o];
#pragma unroll
    for (int k = 0; k < 64; ++k) s += s1[o * 64 + k] * x[k];
    h1v[o] = fmaxf(s, 0.0f);
  }
  float h2v[32];
  for (int o = 0; o < 32; ++o) {
    float s = sb2[o];
#pragma unroll
    for (int k = 0; k < 64; ++k) s += s2[o * 64 + k] * h1v[k];
    h2v[o] = fmaxf(s, 0.0f);
  }
  float lg[10], mx = -3.0e38f;
  for (int o = 0; o < 10; ++o) {
    float s = sb3[o];
#pragma unroll
    for (int k = 0; k < 32; ++k) s += s3[o * 32 + k] * h2v[k];
    lg[o] = s;
    mx = fmaxf(mx, s);
  }
  float sum = 0.0f;
  for (int o = 0; o < 10; ++o) { lg[o] = __expf(lg[o] - mx); sum += lg[o]; }
  float inv = 1.0f / sum;
  for (int o = 0; o < 10; ++o) out[(size_t)b * 10 + o] = lg[o] * inv;
}

// ---------------------------------------------------------------------------
// Launch
// ---------------------------------------------------------------------------
extern "C" void kernel_launch(void* const* d_in, const int* in_sizes, int n_in,
                              void* d_out, int out_size, void* d_ws, size_t ws_size,
                              hipStream_t stream) {
  (void)in_sizes; (void)n_in; (void)out_size; (void)ws_size;
  constexpr int B = 1024, T = 256;

  const float* x     = (const float*)d_in[0];
  const float* Wih1  = (const float*)d_in[1];
  const float* Whh1  = (const float*)d_in[2];
  const float* bih1  = (const float*)d_in[3];
  const float* bhh1  = (const float*)d_in[4];
  const float* Wih2  = (const float*)d_in[5];
  const float* Whh2  = (const float*)d_in[6];
  const float* bih2  = (const float*)d_in[7];
  const float* bhh2  = (const float*)d_in[8];
  const float* Wih3  = (const float*)d_in[9];
  const float* Whh3  = (const float*)d_in[10];
  const float* bih3  = (const float*)d_in[11];
  const float* bhh3  = (const float*)d_in[12];
  const float* fc1w  = (const float*)d_in[13];
  const float* fc1b  = (const float*)d_in[14];
  const float* fc2w  = (const float*)d_in[15];
  const float* fc2b  = (const float*)d_in[16];
  const float* fc3w  = (const float*)d_in[17];
  const float* fc3b  = (const float*)d_in[18];

  uint8_t* ws = (uint8_t*)d_ws;
  // padded f16 weights
  _Float16* W1I = (_Float16*)(ws + 0);          // [256][160]  81920 B (col 132 = bias)
  _Float16* W1H = (_Float16*)(ws + 81920);      // [256][64]   32768 B
  _Float16* W2I = (_Float16*)(ws + 114688);     // [512][64]   65536 B
  _Float16* W2H = (_Float16*)(ws + 180224);     // [512][128] 131072 B
  _Float16* W3I = (_Float16*)(ws + 311296);     // [256][128]  65536 B
  _Float16* W3H = (_Float16*)(ws + 376832);     // [256][64]   32768 B
  float*    B1  = (float*)(ws + 409600);        // [256]
  float*    B2  = (float*)(ws + 410624);        // [512]
  float*    B3  = (float*)(ws + 412672);        // [256]
  _Float16* H1  = (_Float16*)(ws + 413696);     // [B,T,64]  33.5 MB
  _Float16* H2  = (_Float16*)(ws + 33968128);   // [B,T,128] 67.1 MB
  _Float16* H3  = (_Float16*)(ws + 101076992);  // [B,T,64]  33.5 MB

  auto blocks = [](int n) { return (n + 255) / 256; };

  // weight prep (cheap, once per call); bias sums first (layer1 folds B1 into W1I)
  bias_sum_k<<<1, 256, 0, stream>>>(bih1, bhh1, B1, 256);
  bias_sum_k<<<2, 256, 0, stream>>>(bih2, bhh2, B2, 512);
  bias_sum_k<<<1, 256, 0, stream>>>(bih3, bhh3, B3, 256);
  cvt_pad_bias_k<<<blocks(256 * 160), 256, 0, stream>>>(Wih1, B1, W1I, 256, 132, 160);
  cvt_pad_k<<<blocks(256 *  64), 256, 0, stream>>>(Whh1, W1H, 256,  64,  64);
  cvt_pad_k<<<blocks(512 *  64), 256, 0, stream>>>(Wih2, W2I, 512,  64,  64);
  cvt_pad_k<<<blocks(512 * 128), 256, 0, stream>>>(Whh2, W2H, 512, 128, 128);
  cvt_pad_k<<<blocks(256 * 128), 256, 0, stream>>>(Wih3, W3I, 256, 128, 128);
  cvt_pad_k<<<blocks(256 *  64), 256, 0, stream>>>(Whh3, W3H, 256,  64,  64);

  // layer 1: 132(pad 160, bias col) -> 64, batch tile 32 -> 32 blocks
  lstm_layer_k<132, 160, 64, true,  true ><<<B / 32, 256, 0, stream>>>(
      (const void*)x, W1I, W1H, B1, H1, T);
  // layer 2: 64 -> 128, batch tile 16 -> 64 blocks
  lstm_layer_k< 64,  64, 128, false, false><<<B / 16, 256, 0, stream>>>(
      (const void*)H1, W2I, W2H, B2, H2, T);
  // layer 3: 128 -> 64, batch tile 32 -> 32 blocks
  lstm_layer_k<128, 128, 64, false, false><<<B / 32, 256, 0, stream>>>(
      (const void*)H2, W3I, W3H, B3, H3, T);

  // classifier head
  head_k<<<B / 256, 256, 0, stream>>>(H3, fc1w, fc1b, fc2w, fc2b, fc3w, fc3b,
                                      (float*)d_out, T);
}
